// AttentionLayer_78889959293198
// MI455X (gfx1250) — matrix operand order repaired and checked by reading, not compiled, not run on hardware
//
#include <hip/hip_runtime.h>
#include <hip/hip_bf16.h>

// ---------------------------------------------------------------------------
// CDNA5 (gfx1250) fused attention layer.
// B=4096, N=25, D=256, H=8, dk=32, dq=64.
// All GEMMs via v_wmma_f32_16x16x32_bf16 (wave32).
// ---------------------------------------------------------------------------

typedef __attribute__((ext_vector_type(16))) __bf16 v16bf;
typedef __attribute__((ext_vector_type(8)))  float  v8f;

union FragU {
    v16bf  v;
    uint4  u[2];
    __bf16 e[16];
};

__device__ __forceinline__ v8f wmma_bf16(v16bf a, v16bf b, v8f c) {
    // D = A(16x32) * B(32x16) + C(16x16 f32)
    return __builtin_amdgcn_wmma_f32_16x16x32_bf16(
        false, a, false, b, (short)0, c, false, false);
}

// A-matrix fragment (16x32 bf16), row-major source with row stride `stride`
// (elements). Lane L holds row M = L%16. e0..15 -> K = hi*16 + e,
// hi = L>=16 (contiguous, matching the B fragments).
__device__ __forceinline__ v16bf frag_A(const __bf16* base, int row0,
                                        int stride, int k0, int lane) {
    const int r  = row0 + (lane & 15);
    const int hi = lane >> 4;
    const __bf16* p = base + r * stride + k0;
    FragU f;
    f.u[0] = *(const uint4*)(p + hi * 16);
    f.u[1] = *(const uint4*)(p + hi * 16 + 8);
    return f.v;
}

// B-matrix fragment (32x16 bf16) where B[k][c] = base[(col0 + L%16)*stride + k0 + K],
// K = e + hi*16. K is the FAST axis of the source row -> contiguous loads.
__device__ __forceinline__ v16bf frag_B_rowslice(const __bf16* base, int col0,
                                                 int stride, int k0, int lane) {
    const __bf16* p = base + (col0 + (lane & 15)) * stride + k0 + (lane >> 4) * 16;
    FragU f;
    f.u[0] = *(const uint4*)(p);
    f.u[1] = *(const uint4*)(p + 8);
    return f.v;
}

// B-matrix fragment where source is row-major [K][cols]: B[k][c] = base[k*stride + c].
// K = e + hi*16, column fixed per lane -> strided gather (LDS).
__device__ __forceinline__ v16bf frag_B_colmajor(const __bf16* base, int k0,
                                                 int stride, int col0, int lane) {
    const __bf16* p = base + (k0 + (lane >> 4) * 16) * stride + col0 + (lane & 15);
    FragU f;
#pragma unroll
    for (int e = 0; e < 16; ++e) f.e[e] = p[e * stride];
    return f.v;
}

// Pre-swizzled B fragment: w[fragIdx][lane][16] contiguous bf16.
__device__ __forceinline__ v16bf frag_B_swz(const __bf16* w, int fragIdx, int lane) {
    const __bf16* p = w + fragIdx * 512 + lane * 16;
    FragU f;
    f.u[0] = ((const uint4*)p)[0];
    f.u[1] = ((const uint4*)p)[1];
    return f.v;
}

__device__ __forceinline__ float wave_sum(float v) {
#pragma unroll
    for (int off = 16; off > 0; off >>= 1) v += __shfl_xor(v, off, 32);
    return v;
}

__device__ __forceinline__ float tanh_fast(float x) {
    x = fminf(fmaxf(x, -15.0f), 15.0f);
    const float e2 = __expf(2.0f * x);
    return (e2 - 1.0f) / (e2 + 1.0f);
}

// ---------------------------------------------------------------------------
// Workspace layout (bf16 elements): swizzled weight fragments.
// conv weights: [3 taps][8 ktiles][16 ntiles][32 lanes][16 elems]
// ---------------------------------------------------------------------------
#define CONV_SZ   (3 * 8 * 16 * 512)       // 196608 elems per conv weight
#define WQF_SZ    (8 * 4 * 512)            // 16384
#define WO_SZ     (8 * 16 * 512)           // 65536
#define WS_TOTAL  (3 * CONV_SZ + WQF_SZ + WO_SZ)  // 671744 elems

__global__ void prep_weights_kernel(const float* __restrict__ wq,
                                    const float* __restrict__ wk,
                                    const float* __restrict__ wv,
                                    const float* __restrict__ wqf,
                                    const float* __restrict__ wo,
                                    __bf16* __restrict__ ws) {
    const int idx = blockIdx.x * blockDim.x + threadIdx.x;
    if (idx >= WS_TOTAL) return;

    if (idx < 3 * CONV_SZ) {
        const int p    = idx / CONV_SZ;
        const int r    = idx % CONV_SZ;
        const int f    = r >> 9;            // fragment index
        const int lane = (r >> 4) & 31;
        const int e    = r & 15;
        const int t    = f >> 7;            // tap
        const int kk   = (f >> 4) & 7;      // k tile
        const int nn   = f & 15;            // n tile
        const int K    = kk * 32 + ((lane >> 4) << 4) + e;   // input channel i
        const int o    = nn * 16 + (lane & 15);              // output channel
        const float* w = (p == 0) ? wq : (p == 1) ? wk : wv; // [O][I][3]
        ws[idx] = (__bf16)w[o * 768 + K * 3 + t];
    } else if (idx < 3 * CONV_SZ + WQF_SZ) {
        const int r    = idx - 3 * CONV_SZ;
        const int f    = r >> 9;
        const int lane = (r >> 4) & 31;
        const int e    = r & 15;
        const int kk   = f >> 2;
        const int nn   = f & 3;
        const int K    = kk * 32 + ((lane >> 4) << 4) + e;
        const int c    = nn * 16 + (lane & 15);
        ws[idx] = (__bf16)wqf[K * 64 + c];                   // wqf: [256][64]
    } else {
        const int r    = idx - 3 * CONV_SZ - WQF_SZ;
        const int f    = r >> 9;
        const int lane = (r >> 4) & 31;
        const int e    = r & 15;
        const int kk   = f >> 4;
        const int nn   = f & 15;
        const int K    = kk * 32 + ((lane >> 4) << 4) + e;
        const int c    = nn * 16 + (lane & 15);
        ws[idx] = (__bf16)wo[K * 256 + c];                   // wo: [256][256]
    }
}

// ---------------------------------------------------------------------------
// LDS layout (bytes) -- dynamic shared memory
// ---------------------------------------------------------------------------
#define OFF_XPAD 0                     // bf16 [34][256] padded x (row = token+1)
#define OFF_XF32 17408                 // f32  [25][256] x
#define OFF_QBF  43008                 // bf16 [32][256] q  (reused as h after attn*v)
#define OFF_KBF  59392                 // bf16 [32][256] k
#define OFF_VBF  75776                 // bf16 [32][256] v
#define OFF_QF   92160                 // f32  [32][64]  qf
#define OFF_WQP  100352                // f32  [64][8]   wqp
#define OFF_BQP  102400                // f32  [8]       bqp
#define OFF_BQF  102432                // f32  [64]      bqf
#define OFF_SCR  102688                // f32  [32][256] conv scratch / [8][32][32] scores
#define OFF_ATT  135456                // bf16 [8][32][32] attn
#define SMEM_BYTES 151840

__global__ void __launch_bounds__(256, 1)
attn_fused_kernel(const float* __restrict__ x,
                  const float* __restrict__ ln_q_g, const float* __restrict__ ln_q_b,
                  const float* __restrict__ ln_k_g, const float* __restrict__ ln_k_b,
                  const float* __restrict__ ln_v_g, const float* __restrict__ ln_v_b,
                  const float* __restrict__ rel_table,
                  const float* __restrict__ global_bias,
                  const float* __restrict__ alpha,
                  const float* __restrict__ bqf,
                  const float* __restrict__ wqp,
                  const float* __restrict__ bqp,
                  const float* __restrict__ bo,
                  const __bf16* __restrict__ ws,
                  float* __restrict__ out) {
    extern __shared__ char smem[];
    __bf16* xpad   = (__bf16*)(smem + OFF_XPAD);
    float*  xf     = (float*)(smem + OFF_XF32);
    __bf16* qbf    = (__bf16*)(smem + OFF_QBF);
    __bf16* kbf    = (__bf16*)(smem + OFF_KBF);
    __bf16* vbf    = (__bf16*)(smem + OFF_VBF);
    float*  qf_s   = (float*)(smem + OFF_QF);
    float*  wqp_s  = (float*)(smem + OFF_WQP);
    float*  bqp_s  = (float*)(smem + OFF_BQP);
    float*  bqf_s  = (float*)(smem + OFF_BQF);
    float*  scr    = (float*)(smem + OFF_SCR);
    __bf16* att    = (__bf16*)(smem + OFF_ATT);

    const __bf16* wconv_swz[3] = { ws, ws + CONV_SZ, ws + 2 * CONV_SZ };
    const __bf16* wqf_swz = ws + 3 * CONV_SZ;
    const __bf16* wo_swz  = ws + 3 * CONV_SZ + WQF_SZ;

    const int b    = blockIdx.x;
    const int tid  = threadIdx.x;
    const int lane = tid & 31;
    const int wave = tid >> 5;
    const float alphav = alpha[0];

    // ---- Stage 0: load x -> xpad(bf16, zero-padded rows) + xf(f32); params
    const float* xb = x + (size_t)b * 6400;
    for (int i = tid; i < 6400; i += 256) {
        const float v = xb[i];
        const int n = i >> 8, c = i & 255;
        xf[i] = v;
        xpad[(n + 1) * 256 + c] = (__bf16)v;
    }
    for (int i = tid; i < 256; i += 256)       xpad[i] = (__bf16)0.0f;           // row 0
    for (int i = tid; i < 8 * 256; i += 256)   xpad[26 * 256 + i] = (__bf16)0.0f; // rows 26..33
    for (int i = tid; i < 512; i += 256)       wqp_s[i] = wqp[i];
    if (tid < 8)  bqp_s[tid] = bqp[tid];
    if (tid < 64) bqf_s[tid] = bqf[tid];
    __syncthreads();

    // ---- Stage 1: qf = x @ wqf + bqf  (2 Mtiles x 4 Ntiles, K=256) --------
    {
        const int mt = wave >> 2, nt = wave & 3;   // 8 tiles, one per wave
        v8f acc;
#pragma unroll
        for (int e = 0; e < 8; ++e) acc[e] = 0.0f;
#pragma unroll
        for (int kk = 0; kk < 8; ++kk) {
            v16bf a = frag_A(xpad, mt * 16 + 1, 256, kk * 32, lane);
            v16bf bm = frag_B_swz(wqf_swz, kk * 4 + nt, lane);
            acc = wmma_bf16(a, bm, acc);
        }
        const int hi = lane >> 4, c = nt * 16 + (lane & 15);
#pragma unroll
        for (int r = 0; r < 8; ++r)
            qf_s[(mt * 16 + hi * 8 + r) * 64 + c] = acc[r] + bqf_s[c];
    }

    // ---- Stage 2: conv + LayerNorm + residual for q, k, v -----------------
    const float* ln_g[3] = { ln_q_g, ln_k_g, ln_v_g };
    const float* ln_b[3] = { ln_q_b, ln_k_b, ln_v_b };
    __bf16* dst_p[3] = { qbf, kbf, vbf };
#pragma unroll
    for (int p = 0; p < 3; ++p) {
        __syncthreads();                      // SCR free (prev LN done / qf wrote elsewhere)
        const __bf16* wsw = wconv_swz[p];
        for (int tile = wave; tile < 32; tile += 8) {
            const int mt = tile >> 4, nn = tile & 15;
            v8f acc;
#pragma unroll
            for (int e = 0; e < 8; ++e) acc[e] = 0.0f;
#pragma unroll
            for (int t = 0; t < 3; ++t) {
                const int rowbase = mt * 16 + t;   // padded row = token + t (tap offset t-1)
#pragma unroll
                for (int kk = 0; kk < 8; ++kk) {
                    v16bf a = frag_A(xpad, rowbase, 256, kk * 32, lane);
                    v16bf bm = frag_B_swz(wsw, (t * 8 + kk) * 16 + nn, lane);
                    acc = wmma_bf16(a, bm, acc);
                }
            }
            const int hi = lane >> 4, c = nn * 16 + (lane & 15);
#pragma unroll
            for (int r = 0; r < 8; ++r)
                scr[(mt * 16 + hi * 8 + r) * 256 + c] = acc[r];
        }
        __syncthreads();
        // LayerNorm + residual, wave-per-row
        const float* g = ln_g[p];
        const float* bb = ln_b[p];
        __bf16* dst = dst_p[p];
        for (int r = wave; r < 32; r += 8) {
            if (r < 25) {
                float s = 0.0f, ss = 0.0f;
#pragma unroll
                for (int j = 0; j < 8; ++j) {
                    const float v = scr[r * 256 + lane + 32 * j];
                    s += v; ss += v * v;
                }
                s = wave_sum(s); ss = wave_sum(ss);
                const float mu  = s * (1.0f / 256.0f);
                const float var = ss * (1.0f / 256.0f) - mu * mu;
                const float rs  = rsqrtf(var + 1e-5f);
#pragma unroll
                for (int j = 0; j < 8; ++j) {
                    const int c = lane + 32 * j;
                    const float v = (scr[r * 256 + c] - mu) * rs * g[c] + bb[c]
                                    + xf[r * 256 + c];
                    dst[r * 256 + c] = (__bf16)v;
                }
            } else {
#pragma unroll
                for (int j = 0; j < 8; ++j)
                    dst[r * 256 + lane + 32 * j] = (__bf16)0.0f;
            }
        }
    }
    __syncthreads();

    // ---- Stage 3: scores = QK^T/sqrt(dk) + rel + global*alpha -------------
    // scr reused as scores f32 [8][32][32]
    for (int tile = wave; tile < 32; tile += 8) {
        const int h = tile >> 2, mt = (tile >> 1) & 1, nt = tile & 1;
        v16bf a = frag_A(qbf, mt * 16, 256, h * 32, lane);
        v16bf bm = frag_B_rowslice(kbf, nt * 16, 256, h * 32, lane);
        v8f acc;
#pragma unroll
        for (int e = 0; e < 8; ++e) acc[e] = 0.0f;
        acc = wmma_bf16(a, bm, acc);
        const float scale = 0.17677669529663687f;  // 1/sqrt(32)
        const int hi = lane >> 4, m = nt * 16 + (lane & 15);
#pragma unroll
        for (int r = 0; r < 8; ++r) {
            const int n = mt * 16 + hi * 8 + r;
            float val = acc[r] * scale;
            if (n < 25 && m < 25)
                val += rel_table[(n - m + 24) * 8 + h]
                     + global_bias[h * 625 + n * 25 + m] * alphav;
            scr[h * 1024 + n * 32 + m] = val;
        }
    }
    __syncthreads();

    // ---- Stage 4: dynamic adjacency bias: tanh(qf_n - qf_m) @ wqp + bqp ---
    for (int pm = tid; pm < 625; pm += 256) {
        const int n = pm / 25, m = pm % 25;
        float acc[8];
#pragma unroll
        for (int h = 0; h < 8; ++h) acc[h] = 0.0f;
        for (int c = 0; c < 64; ++c) {
            const float t = tanh_fast(qf_s[n * 64 + c] - qf_s[m * 64 + c]);
#pragma unroll
            for (int h = 0; h < 8; ++h) acc[h] += t * wqp_s[c * 8 + h];
        }
#pragma unroll
        for (int h = 0; h < 8; ++h)
            scr[h * 1024 + n * 32 + m] += acc[h] + bqp_s[h];
    }
    __syncthreads();

    // ---- Stage 5: softmax over m, convert to bf16 (pad cols/rows zero) ----
    if (tid < 200) {
        const int h = tid / 25, n = tid % 25;
        float* row = scr + h * 1024 + n * 32;
        float mx = -1e30f;
        for (int m = 0; m < 25; ++m) mx = fmaxf(mx, row[m]);
        float s = 0.0f;
        for (int m = 0; m < 25; ++m) { row[m] = __expf(row[m] - mx); s += row[m]; }
        const float inv = 1.0f / s;
        __bf16* arow = att + h * 1024 + n * 32;
        for (int m = 0; m < 25; ++m) arow[m] = (__bf16)(row[m] * inv);
        for (int m = 25; m < 32; ++m) arow[m] = (__bf16)0.0f;
    } else if (tid - 200 < 56) {
        const int r = tid - 200;
        const int h = r / 7, n = 25 + r % 7;
        __bf16* arow = att + h * 1024 + n * 32;
        for (int m = 0; m < 32; ++m) arow[m] = (__bf16)0.0f;
    }
    __syncthreads();

    // ---- Stage 6: hout = attn @ V  (per head, K = 32 padded) --------------
    // writes into qbf (reused as h buffer; q no longer needed)
    for (int tile = wave; tile < 32; tile += 8) {
        const int h = tile >> 2, mt = (tile >> 1) & 1, nt = tile & 1;
        v16bf a = frag_A(att + h * 1024, mt * 16, 32, 0, lane);
        v16bf bm = frag_B_colmajor(vbf, 0, 256, h * 32 + nt * 16, lane);
        v8f acc;
#pragma unroll
        for (int e = 0; e < 8; ++e) acc[e] = 0.0f;
        acc = wmma_bf16(a, bm, acc);
        const int hi = lane >> 4, d = h * 32 + nt * 16 + (lane & 15);
#pragma unroll
        for (int r = 0; r < 8; ++r)
            qbf[(mt * 16 + hi * 8 + r) * 256 + d] = (__bf16)acc[r];
    }
    __syncthreads();

    // ---- Stage 7: out = hout @ wo + bo ------------------------------------
    float* ob = out + (size_t)b * 6400;
    for (int tile = wave; tile < 32; tile += 8) {
        const int mt = tile >> 4, nn = tile & 15;
        v8f acc;
#pragma unroll
        for (int e = 0; e < 8; ++e) acc[e] = 0.0f;
#pragma unroll
        for (int kk = 0; kk < 8; ++kk) {
            v16bf a = frag_A(qbf, mt * 16, 256, kk * 32, lane);
            v16bf bm = frag_B_swz(wo_swz, kk * 16 + nn, lane);
            acc = wmma_bf16(a, bm, acc);
        }
        const int hi = lane >> 4, c = nn * 16 + (lane & 15);
#pragma unroll
        for (int r = 0; r < 8; ++r) {
            const int n = mt * 16 + hi * 8 + r;
            if (n < 25) ob[n * 256 + c] = acc[r] + bo[c];
        }
    }
}

// ---------------------------------------------------------------------------
extern "C" void kernel_launch(void* const* d_in, const int* in_sizes, int n_in,
                              void* d_out, int out_size, void* d_ws, size_t ws_size,
                              hipStream_t stream) {
    const float* x           = (const float*)d_in[0];
    const float* wq          = (const float*)d_in[1];
    const float* wk          = (const float*)d_in[2];
    const float* wv          = (const float*)d_in[3];
    const float* ln_q_g      = (const float*)d_in[4];
    const float* ln_q_b      = (const float*)d_in[5];
    const float* ln_k_g      = (const float*)d_in[6];
    const float* ln_k_b      = (const float*)d_in[7];
    const float* ln_v_g      = (const float*)d_in[8];
    const float* ln_v_b      = (const float*)d_in[9];
    const float* rel_table   = (const float*)d_in[10];
    const float* global_bias = (const float*)d_in[11];
    const float* alpha       = (const float*)d_in[12];
    const float* wqf         = (const float*)d_in[13];
    const float* bqf         = (const float*)d_in[14];
    const float* wqp         = (const float*)d_in[15];
    const float* bqp         = (const float*)d_in[16];
    const float* wo          = (const float*)d_in[17];
    const float* bo          = (const float*)d_in[18];
    float* out = (float*)d_out;
    __bf16* ws = (__bf16*)d_ws;

    (void)in_sizes; (void)n_in; (void)out_size; (void)ws_size;

    // allow >64KB dynamic LDS (benign outside capture semantics; not a stream op)
    (void)hipFuncSetAttribute((const void*)attn_fused_kernel,
                              hipFuncAttributeMaxDynamicSharedMemorySize,
                              SMEM_BYTES);

    const int prep_blocks = (WS_TOTAL + 255) / 256;
    prep_weights_kernel<<<prep_blocks, 256, 0, stream>>>(wq, wk, wv, wqf, wo, ws);

    attn_fused_kernel<<<4096, 256, SMEM_BYTES, stream>>>(
        x, ln_q_g, ln_q_b, ln_k_g, ln_k_b, ln_v_g, ln_v_b,
        rel_table, global_bias, alpha, bqf, wqp, bqp, bo, ws, out);
}
